// SoftArgmaxCorrespondence_10136122819255
// MI455X (gfx1250) — matrix operand order repaired and checked by reading, not compiled, not run on hardware
//
#include <hip/hip_runtime.h>

#define B_    4
#define C_    256
#define HW_   4096
#define W_    64
#define EPSN  1e-12f
#define SCALE 0.625f   // (1/sqrt(256)) * (1/0.1)

typedef __attribute__((ext_vector_type(16))) _Float16 v16h;
typedef __attribute__((ext_vector_type(8)))  float    v8f;

#if __has_builtin(__builtin_amdgcn_global_load_async_to_lds_b128) && \
    __has_builtin(__builtin_amdgcn_s_wait_asynccnt)
#define HAS_ASYNC_LDS 1
typedef __attribute__((ext_vector_type(4))) int v4i_;
typedef __attribute__((address_space(1))) v4i_ gv4i;   // global int4*
typedef __attribute__((address_space(3))) v4i_ lv4i;   // LDS int4*
#else
#define HAS_ASYNC_LDS 0
#endif

// ---------------------------------------------------------------------------
// Kernel 1: L2-normalize feat_b along C, convert to f16, keep (B, C, HW) layout
// (already K-major for the WMMA B-matrix side). grid(HW/256, B), 256 thr.
// ---------------------------------------------------------------------------
__global__ __launch_bounds__(256) void norm_b_kernel(const float* __restrict__ fb,
                                                     _Float16* __restrict__ fbN) {
  int b   = blockIdx.y;
  int pix = blockIdx.x * 256 + threadIdx.x;
  const float* src = fb + (size_t)b * C_ * HW_ + pix;
  float ss = 0.f;
  for (int c = 0; c < C_; ++c) {
    float v = src[(size_t)c * HW_];
    ss += v * v;
  }
  float rn = 1.0f / fmaxf(sqrtf(ss), EPSN);
  _Float16* dst = fbN + (size_t)b * C_ * HW_ + pix;
  for (int c = 0; c < C_; ++c)
    dst[(size_t)c * HW_] = (_Float16)(src[(size_t)c * HW_] * rn);
}

// ---------------------------------------------------------------------------
// Kernel 2: L2-normalize feat_a along C, transpose to (B, HW, C) f16 (WMMA
// A-matrix side, M-major rows). LDS-tiled transpose. grid(HW/64, B), 256 thr.
// ---------------------------------------------------------------------------
__global__ __launch_bounds__(256) void norm_a_kernel(const float* __restrict__ fa,
                                                     _Float16* __restrict__ faT) {
  __shared__ float red[4][64];
  __shared__ float rnorm[64];
  __shared__ float tile[64][65];   // +1 pad: conflict-free transposed reads
  int b     = blockIdx.y;
  int pBase = blockIdx.x * 64;
  int tp = threadIdx.x & 63;       // pixel within tile
  int tc = threadIdx.x >> 6;       // 0..3
  const float* src = fa + (size_t)b * C_ * HW_ + pBase;

  float ss = 0.f;
  for (int c = tc; c < C_; c += 4) {
    float v = src[(size_t)c * HW_ + tp];
    ss += v * v;
  }
  red[tc][tp] = ss;
  __syncthreads();
  if (threadIdx.x < 64) {
    float s = red[0][threadIdx.x] + red[1][threadIdx.x] +
              red[2][threadIdx.x] + red[3][threadIdx.x];
    rnorm[threadIdx.x] = 1.0f / fmaxf(sqrtf(s), EPSN);
  }
  __syncthreads();

  for (int cb = 0; cb < 4; ++cb) {
    for (int pass = 0; pass < 16; ++pass) {
      int cl = tc * 16 + pass;  // 0..63
      tile[cl][tp] = src[(size_t)(cb * 64 + cl) * HW_ + tp] * rnorm[tp];
    }
    __syncthreads();
    int co = threadIdx.x & 63;
    int po = threadIdx.x >> 6;
    for (int pass = 0; pass < 16; ++pass) {
      int p = po + pass * 4;
      faT[((size_t)b * HW_ + pBase + p) * C_ + cb * 64 + co] = (_Float16)tile[co][p];
    }
    __syncthreads();
  }
}

// ---------------------------------------------------------------------------
// Kernel 3: fused corr-GEMM (f16 WMMA) + bounded softmax + soft-argmax flow.
// grid(HW/128, B), block 256 (8 waves); wave w owns 16 rows (128 rows/block).
// B chunks double-buffered in LDS via async global->LDS copies (ASYNCcnt),
// overlapped with the WMMA chains. Never materializes the 4096x4096 corr.
// ---------------------------------------------------------------------------
#define BROW_STRIDE 72   // 64 cols + 8 pad halves -> 144B rows (16B aligned)
#define NCHUNK      (HW_ / 64)

__global__ __launch_bounds__(256) void flow_kernel(const _Float16* __restrict__ faT,
                                                   const _Float16* __restrict__ fbN,
                                                   float* __restrict__ out) {
  __shared__ __align__(16) _Float16 Btile[2][256 * BROW_STRIDE];  // 2 x 36 KB

  int b     = blockIdx.y;
  int iBase = blockIdx.x * 128;
  int t     = threadIdx.x;
  int lane  = t & 31;
  int wave  = t >> 5;   // 0..7

  // Preload this wave's A fragments (16 rows x full K=256) from global once.
  // A 16-bit 16x32 layout: lane&15 = M; halves 0-7/8-15 = K {0..7,16..23}
  // for lanes 0-15, {8..15,24..31} for lanes 16-31.
  int m  = lane & 15;
  int k0 = (lane < 16) ? 0 : 8;
  const _Float16* aRowPtr = faT + ((size_t)b * HW_ + iBase + wave * 16 + m) * C_;
  v16h afrag[8];
#pragma unroll
  for (int kk = 0; kk < 8; ++kk) {
    union { uint4 u[2]; v16h v; } ua;
    const uint4* pa = (const uint4*)(aRowPtr + kk * 32 + k0);
    ua.u[0] = pa[0];
    ua.u[1] = pa[2];   // +16 halves
    afrag[kk] = ua.v;
  }

  float sS[8], sX[8], sY[8];
#pragma unroll
  for (int r = 0; r < 8; ++r) { sS[r] = 0.f; sX[r] = 0.f; sY[r] = 0.f; }

  const _Float16* fbB = fbN + (size_t)b * C_ * HW_;

  // Stage one 256x64 B chunk (cols j0..j0+63) into LDS buffer `buf`.
  // 2048 16-byte segments, 8 per thread.
  auto stage = [&](int bufsel, int j0) {
    _Float16* dst = &Btile[bufsel][0];
    for (int k = t; k < 2048; k += 256) {
      int row = k >> 3;   // c
      int seg = k & 7;    // 8 halves per segment
      const _Float16* g = fbB + (size_t)row * HW_ + j0 + seg * 8;
      _Float16* l = dst + row * BROW_STRIDE + seg * 8;
#if HAS_ASYNC_LDS
      __builtin_amdgcn_global_load_async_to_lds_b128(
          (gv4i*)g, (lv4i*)l, 0, 0);
#else
      *(uint4*)l = *(const uint4*)g;
#endif
    }
  };

  stage(0, 0);
#if HAS_ASYNC_LDS
  __builtin_amdgcn_s_wait_asynccnt(0);
#endif
  __syncthreads();

  for (int jc = 0; jc < NCHUNK; ++jc) {
    int j0 = jc * 64;
    const _Float16* Bcur = &Btile[jc & 1][0];

    // Kick off async staging of the next chunk; overlaps with WMMA below.
    if (jc + 1 < NCHUNK) {
      stage((jc + 1) & 1, j0 + 64);
#if !HAS_ASYNC_LDS
      __builtin_prefetch(fbB + j0 + 128, 0, 0);
#endif
    }

#pragma unroll
    for (int sub = 0; sub < 4; ++sub) {
      v8f acc = {};
#pragma unroll
      for (int kk = 0; kk < 8; ++kk) {
        // B 16-bit 32x16 layout: lane&15 = N; halves K {0..7,16..23} for lanes 0-15, {8..15,24..31} for lanes 16-31
        union { uint4 u[2]; v16h v; } ub;
        const _Float16* pb = Bcur + (kk * 32 + ((lane >> 4) << 3)) * BROW_STRIDE + sub * 16 + (lane & 15);
#pragma unroll
        for (int s = 0; s < 16; ++s)
          ub.v[s] = pb[((s & 7) + ((s >> 3) << 4)) * BROW_STRIDE];
        acc = __builtin_amdgcn_wmma_f32_16x16x32_f16(
            false, afrag[kk], false, ub.v, (short)0, acc, false, false);
      }
      // C/D layout: VGPR r -> row (r or r+8 by half-wave), lane&15 -> col
      int col  = j0 + sub * 16 + (lane & 15);
      float xj = (float)(col & (W_ - 1));
      float yj = (float)(col >> 6);
#pragma unroll
      for (int r = 0; r < 8; ++r) {
        float e = __expf(acc[r] * SCALE);  // |logit| <= 0.625: no max needed
        sS[r] += e;
        sX[r] += e * xj;
        sY[r] += e * yj;
      }
    }

#if HAS_ASYNC_LDS
    __builtin_amdgcn_s_wait_asynccnt(0);
#endif
    __syncthreads();   // next buffer fully staged & visible before reuse
  }

  // Horizontal reduce across the 16 lanes of each half-wave (xor 8,4,2,1
  // stays within a 16-lane half), then lanes 0 / 16 emit rows r / r+8.
#pragma unroll
  for (int r = 0; r < 8; ++r) {
    float s = sS[r], x = sX[r], y = sY[r];
    for (int msk = 8; msk > 0; msk >>= 1) {
      s += __shfl_xor(s, msk, 32);
      x += __shfl_xor(x, msk, 32);
      y += __shfl_xor(y, msk, 32);
    }
    if ((lane & 15) == 0) {
      int row  = iBase + wave * 16 + r + ((lane >> 4) << 3);
      float xi = (float)(row & (W_ - 1));
      float yi = (float)(row >> 6);
      float inv = 1.0f / s;
      out[(size_t)b * 2 * HW_ + row]       = x * inv - xi;  // flow x
      out[(size_t)b * 2 * HW_ + HW_ + row] = y * inv - yi;  // flow y
    }
  }
}

// ---------------------------------------------------------------------------
extern "C" void kernel_launch(void* const* d_in, const int* in_sizes, int n_in,
                              void* d_out, int out_size, void* d_ws, size_t ws_size,
                              hipStream_t stream) {
  const float* fa = (const float*)d_in[0];
  const float* fb = (const float*)d_in[1];
  float* out = (float*)d_out;

  _Float16* faT = (_Float16*)d_ws;                              // 8 MB (B,HW,C)
  _Float16* fbN = (_Float16*)((char*)d_ws +
                              (size_t)B_ * HW_ * C_ * sizeof(_Float16));  // 8 MB (B,C,HW)

  norm_a_kernel<<<dim3(HW_ / 64, B_), 256, 0, stream>>>(fa, faT);
  norm_b_kernel<<<dim3(HW_ / 256, B_), 256, 0, stream>>>(fb, fbN);
  flow_kernel<<<dim3(HW_ / 128, B_), 256, 0, stream>>>(faT, fbN, out);
}
